// GNNModel_53154515255971
// MI455X (gfx1250) — matrix operand order repaired and checked by reading, not compiled, not run on hardware
//
#include <hip/hip_runtime.h>
#include <hip/hip_bf16.h>
#include <hip/hip_fp16.h>

// GNN forward on MI455X (gfx1250): all linear layers via v_wmma_f32_16x16x32_f16
// (f16 data, f32 accumulate). Weight tile staged in LDS ONCE per workgroup
// (K<=512 -> <=65KB of the WGP's 320KB LDS), so the K-loop is barrier-free:
// global_load_b128 (A) + ds_load_b128 (B) + v_wmma only.

typedef _Float16 h16;
typedef _Float16 v8h  __attribute__((ext_vector_type(8)));
typedef _Float16 v16h __attribute__((ext_vector_type(16)));
typedef float    v8f  __attribute__((ext_vector_type(8)));

#define NN 10000
#define EE 64000
#define BB 4
#define KMAX 512

static __device__ __forceinline__ float warp_sum32(float v) {
#pragma unroll
  for (int off = 16; off > 0; off >>= 1) v += __shfl_xor(v, off, 32);
  return v;
}

// act: 0=none 1=relu 2=sigmoid 3=elu 4=leaky(0.01)
static __device__ __forceinline__ float act_apply(float v, int act) {
  if (act == 1) return v > 0.f ? v : 0.f;
  if (act == 2) return 1.f / (1.f + __expf(-v));
  if (act == 3) return v > 0.f ? v : (__expf(v) - 1.f);
  if (act == 4) return v > 0.f ? v : 0.01f * v;
  return v;
}

// ---------------------------------------------------------------- utilities
__global__ void k_zero(float* p, int n) {
  int i = blockIdx.x * blockDim.x + threadIdx.x;
  if (i < n) p[i] = 0.f;
}

__global__ void k_f32_to_f16(const float* __restrict__ in, h16* __restrict__ out, int n) {
  int i = blockIdx.x * blockDim.x + threadIdx.x;
  if (i < n) out[i] = (h16)in[i];
}

// W [K,N] f32 row-major  ->  Wt [N,K] f16 (so B-fragment reads are contiguous)
__global__ void k_convert_weight_t(const float* __restrict__ W, h16* __restrict__ Wt,
                                   int K, int N) {
  int idx = blockIdx.x * blockDim.x + threadIdx.x;
  if (idx >= K * N) return;
  int k = idx / N, n = idx % N;
  Wt[(size_t)n * K + k] = (h16)W[idx];
}

// ---------------------------------------------------------------- layernorm (wave per row)
__global__ void k_layernorm_f16(const float* __restrict__ x, const float* __restrict__ g,
                                const float* __restrict__ bt, h16* __restrict__ out,
                                int rows, int D) {
  int wave = threadIdx.x >> 5, lane = threadIdx.x & 31;
  int row = blockIdx.x * 8 + wave;
  if (row >= rows) return;
  const float* xr = x + (size_t)row * D;
  float vals[8];
  int cnt = D >> 5;
  float s = 0.f;
  for (int i = 0; i < cnt; ++i) { vals[i] = xr[lane + i * 32]; s += vals[i]; }
  s = warp_sum32(s);
  float mu = s / (float)D;
  float var = 0.f;
  for (int i = 0; i < cnt; ++i) { float d = vals[i] - mu; var += d * d; }
  var = warp_sum32(var) / (float)D;
  float rs = rsqrtf(var + 1e-5f);
  h16* orow = out + (size_t)row * D;
  for (int i = 0; i < cnt; ++i) {
    int c = lane + i * 32;
    orow[c] = (h16)(g[c] * (vals[i] - mu) * rs + bt[c]);
  }
}

// divide by deg, layernorm, activation; optional f32/f16 outputs with row strides
__global__ void k_mean_ln_act(const float* __restrict__ acc, const float* __restrict__ deg,
                              const float* __restrict__ g, const float* __restrict__ bt,
                              float* out32, int ld32, h16* out16, int ld16,
                              int rows, int D, int act) {
  int wave = threadIdx.x >> 5, lane = threadIdx.x & 31;
  int row = blockIdx.x * 8 + wave;
  if (row >= rows) return;
  float inv = 1.f / fmaxf(deg[row], 1.f);
  const float* ar = acc + (size_t)row * D;
  float vals[8];
  int cnt = D >> 5;
  float s = 0.f;
  for (int i = 0; i < cnt; ++i) { vals[i] = ar[lane + i * 32] * inv; s += vals[i]; }
  s = warp_sum32(s);
  float mu = s / (float)D;
  float var = 0.f;
  for (int i = 0; i < cnt; ++i) { float d = vals[i] - mu; var += d * d; }
  var = warp_sum32(var) / (float)D;
  float rs = rsqrtf(var + 1e-5f);
  for (int i = 0; i < cnt; ++i) {
    int c = lane + i * 32;
    float v = act_apply(g[c] * (vals[i] - mu) * rs + bt[c], act);
    if (out32) out32[(size_t)row * ld32 + c] = v;
    if (out16) out16[(size_t)row * ld16 + c] = (h16)v;
  }
}

// divide by deg only (emean) with strided f32 + f16 outputs
__global__ void k_mean_scale(const float* __restrict__ acc, const float* __restrict__ deg,
                             float* __restrict__ out32, int ld32,
                             h16* __restrict__ out16, int ld16, int rows, int D) {
  int idx = blockIdx.x * blockDim.x + threadIdx.x;
  if (idx >= rows * D) return;
  int r = idx / D, c = idx % D;
  float v = acc[idx] / fmaxf(deg[r], 1.f);
  out32[(size_t)r * ld32 + c] = v;
  out16[(size_t)r * ld16 + c] = (h16)v;
}

// ---------------------------------------------------------------- WMMA GEMM
// C[M,N] = act(A[M,K](f16) @ W[K,N] + bias), W given transposed: Wt[N,K] f16.
// Block = 256 threads = 8 waves; tile 128(M) x 64(N).
// The 64-col weight tile for ALL of K is staged in LDS once (K<=512 -> 65KB),
// then the K-loop runs with no barriers. Row stride K+8 f16 => per-lane LDS
// stride = 4 banks (mod 64) -> conflict-free b128 B-fragment reads.
__global__ void __launch_bounds__(256)
k_gemm_wmma(const h16* __restrict__ A, const h16* __restrict__ Wt,
            const float* __restrict__ bias, float* __restrict__ Cf,
            h16* __restrict__ Ch, int M, int N, int K, int act) {
  __shared__ h16 ldsB[64 * (KMAX + 8)];
  const int tid  = threadIdx.x;
  const int wave = tid >> 5;
  const int lane = tid & 31;
  const int kh   = lane >> 4;   // half-select (0/1)
  const int l16  = lane & 15;
  const int mbase = blockIdx.x * 128 + wave * 16;
  const int nbase = blockIdx.y * 64;
  const int ldb = K + 8;

  // cooperative one-shot weight-tile load: 64 cols x K, v8h (16B) chunks
  const int chunksPerCol = K >> 3;
  const int totalChunks = 64 * chunksPerCol;
  for (int id = tid; id < totalChunks; id += 256) {
    int c  = id / chunksPerCol;
    int ko = (id - c * chunksPerCol) * 8;
    *(v8h*)(&ldsB[c * ldb + ko]) = *(const v8h*)(Wt + (size_t)(nbase + c) * K + ko);
  }
  __syncthreads();

  int arow = mbase + l16;
  if (arow >= M) arow = M - 1;  // clamp; stores are guarded
  const h16* Arow = A + (size_t)arow * K;

  v8f acc[4] = {};

  for (int kk = 0; kk < K; kk += 32) {
    // keep the streaming A operand ahead of us (global_prefetch_b8)
    __builtin_prefetch(Arow + kk + 256, 0, 1);
    // A fragment (16x32 f16): lane half kh holds K in [8kh,8kh+8) and [16+8kh,...)
    v8h alo = *(const v8h*)(Arow + kk + kh * 8);
    v8h ahi = *(const v8h*)(Arow + kk + 16 + kh * 8);
    v16h afrag = __builtin_shufflevector(alo, ahi, 0, 1, 2, 3, 4, 5, 6, 7,
                                         8, 9, 10, 11, 12, 13, 14, 15);
#pragma unroll
    for (int t = 0; t < 4; ++t) {
      // B fragment (32x16 f16): lane half kh holds K in [8kh,8kh+8) and [16+8kh,...) of column l16
      const h16* bp = &ldsB[(t * 16 + l16) * ldb + kk + kh * 8];
      v8h blo = *(const v8h*)(bp);
      v8h bhi = *(const v8h*)(bp + 16);
      v16h bfrag = __builtin_shufflevector(blo, bhi, 0, 1, 2, 3, 4, 5, 6, 7,
                                           8, 9, 10, 11, 12, 13, 14, 15);
      acc[t] = __builtin_amdgcn_wmma_f32_16x16x32_f16(
          false, afrag, false, bfrag, (short)0, acc[t], false, false);
    }
  }

  // epilogue: D layout -> VGPR r holds M=r+8*kh, N = l16 (+ tile offsets)
#pragma unroll
  for (int t = 0; t < 4; ++t) {
    int n = nbase + t * 16 + l16;
    float bv = bias ? bias[n] : 0.f;
#pragma unroll
    for (int r = 0; r < 8; ++r) {
      int m = mbase + r + kh * 8;
      if (m < M) {
        float v = act_apply(acc[t][r] + bv, act);
        size_t off = (size_t)m * N + n;
        if (Cf) Cf[off] = v;
        if (Ch) Ch[off] = (h16)v;
      }
    }
  }
}

// ---------------------------------------------------------------- small heads (N=1)
__global__ void k_rowdot(const h16* __restrict__ A, const float* __restrict__ w,
                         const float* __restrict__ bias, float* __restrict__ out,
                         int M, int K, int dosig) {
  int r = blockIdx.x * blockDim.x + threadIdx.x;
  if (r >= M) return;
  const h16* a = A + (size_t)r * K;
  float s = bias[0];
  for (int k = 0; k < K; ++k) s += (float)a[k] * w[k];
  out[r] = dosig ? 1.f / (1.f + __expf(-s)) : s;
}

// ---------------------------------------------------------------- edge/graph ops
__global__ void k_mul_w(float* __restrict__ e3, h16* __restrict__ ench,
                        const float* __restrict__ wc, int E_, int C) {
  int idx = blockIdx.x * blockDim.x + threadIdx.x;
  if (idx >= E_ * C) return;
  int e = idx / C;
  float v = e3[idx] * wc[e];
  e3[idx] = v;
  ench[idx] = (h16)v;
}

// vectorized gather: one 16-byte (v8h) chunk per thread
__global__ void k_gather_concat(const h16* __restrict__ xh, int D,
                                const h16* __restrict__ ench, int C,
                                const int* __restrict__ src, const int* __restrict__ dst,
                                h16* __restrict__ out, int E_) {
  int W = 2 * D + C;
  int Wc = W >> 3;
  int idx = blockIdx.x * blockDim.x + threadIdx.x;
  if (idx >= E_ * Wc) return;
  int e = idx / Wc, j = (idx - e * Wc) << 3;
  v8h v;
  if (j < D)          v = *(const v8h*)(xh + (size_t)dst[e] * D + j);        // x_i
  else if (j < 2 * D) v = *(const v8h*)(xh + (size_t)src[e] * D + (j - D));  // x_j
  else                v = *(const v8h*)(ench + (size_t)e * C + (j - 2 * D));
  *(v8h*)(out + (size_t)e * W + j) = v;
}

__global__ void k_deg(const int* __restrict__ dst, float* __restrict__ deg, int E_) {
  int e = blockIdx.x * blockDim.x + threadIdx.x;
  if (e < E_) atomicAdd(&deg[dst[e]], 1.0f);
}

__global__ void k_scatter_add(const float* __restrict__ vals, const int* __restrict__ dst,
                              float* __restrict__ acc, int E_, int C) {
  int idx = blockIdx.x * blockDim.x + threadIdx.x;
  if (idx >= E_ * C) return;
  int e = idx / C, c = idx - e * C;
  atomicAdd(&acc[(size_t)dst[e] * C + c], vals[idx]);
}

__global__ void k_xf_blend(const float* __restrict__ skip, const float* __restrict__ gate,
                           const float* __restrict__ x2, float* __restrict__ out32, int ld32,
                           h16* __restrict__ out16, int ld16, int rows, int C) {
  int idx = blockIdx.x * blockDim.x + threadIdx.x;
  if (idx >= rows * C) return;
  int r = idx / C, c = idx - r * C;
  float g = gate[idx];
  float v = g * skip[idx] + (1.f - g) * x2[idx];
  out32[(size_t)r * ld32 + c] = v;
  out16[(size_t)r * ld16 + c] = (h16)v;
}

// ---------------------------------------------------------------- launch
extern "C" void kernel_launch(void* const* d_in, const int* in_sizes, int n_in,
                              void* d_out, int out_size, void* d_ws, size_t ws_size,
                              hipStream_t stream) {
  (void)in_sizes; (void)n_in; (void)out_size; (void)ws_size;

  // ---- inputs (params flattened in sorted dict-key order, W before b) ----
  const float* x_in = (const float*)d_in[0];
  const int*   ei   = (const int*)d_in[1];
  const float* ea   = (const float*)d_in[2];
  const float* cls1W = (const float*)d_in[3];  const float* cls1b = (const float*)d_in[4];
  const float* cls2W = (const float*)d_in[5];  const float* cls2b = (const float*)d_in[6];
  const float* c11W = (const float*)d_in[7];   const float* c11b = (const float*)d_in[8];
  const float* c12W = (const float*)d_in[9];   const float* c12b = (const float*)d_in[10];
  const float* c13W = (const float*)d_in[11];  const float* c13b = (const float*)d_in[12];
  const float* c21W = (const float*)d_in[13];  const float* c21b = (const float*)d_in[14];
  const float* c22W = (const float*)d_in[15];  const float* c22b = (const float*)d_in[16];
  const float* c23W = (const float*)d_in[17];  const float* c23b = (const float*)d_in[18];
  const float* enc1W = (const float*)d_in[19]; const float* enc1b = (const float*)d_in[20];
  const float* enc2W = (const float*)d_in[21]; const float* enc2b = (const float*)d_in[22];
  const float* enc3W = (const float*)d_in[23]; const float* enc3b = (const float*)d_in[24];
  const float* elng = (const float*)d_in[25];  const float* elnb = (const float*)d_in[26];
  const float* gateW = (const float*)d_in[27]; const float* gateb = (const float*)d_in[28];
  const float* ln0g = (const float*)d_in[29];  const float* ln0b = (const float*)d_in[30];
  const float* ln1g = (const float*)d_in[31];  const float* ln1b = (const float*)d_in[32];
  const float* ln2g = (const float*)d_in[33];  const float* ln2b = (const float*)d_in[34];
  const float* np1W = (const float*)d_in[35];  const float* np1b = (const float*)d_in[36];
  const float* np2W = (const float*)d_in[37];  const float* np2b = (const float*)d_in[38];
  const float* np3W = (const float*)d_in[39];  const float* np3b = (const float*)d_in[40];
  const float* skipW = (const float*)d_in[41]; const float* skipb = (const float*)d_in[42];

  float* out = (float*)d_out;
  float* p_base = out + (size_t)BB * NN * 512;

  // ---- workspace bump allocator ----
  char* wp = (char*)d_ws;
  auto alloc = [&](size_t bytes) -> void* {
    void* r = (void*)wp; wp += (bytes + 255) & ~(size_t)255; return r;
  };
  // f16 activation buffers
  h16* xln16 = (h16*)alloc((size_t)NN * 128 * 2);
  h16* ea16  = (h16*)alloc((size_t)EE * 64 * 2);
  h16* ealn16= (h16*)alloc((size_t)EE * 64 * 2);
  h16* ecls  = (h16*)alloc((size_t)EE * 64 * 2);
  h16* e1h   = (h16*)alloc((size_t)EE * 256 * 2);
  h16* mbuf  = (h16*)alloc((size_t)EE * 512 * 2);  // enc2 out, then conv gather input
  h16* eench = (h16*)alloc((size_t)EE * 256 * 2);
  h16* c1h   = (h16*)alloc((size_t)EE * 128 * 2);
  h16* c2h   = (h16*)alloc((size_t)EE * 256 * 2);
  h16* x1h   = (h16*)alloc((size_t)NN * 128 * 2);
  h16* skiph = (h16*)alloc((size_t)NN * 256 * 2);
  h16* xfh   = (h16*)alloc((size_t)NN * 512 * 2);
  h16* h1h   = (h16*)alloc((size_t)NN * 256 * 2);
  h16* h2h   = (h16*)alloc((size_t)NN * 128 * 2);
  // f32 buffers
  float* F1    = (float*)alloc((size_t)EE * 256 * 4);  // e3 / e_enc (f32)
  float* F2    = (float*)alloc((size_t)EE * 256 * 4);  // conv layer-3 outputs
  float* wcls  = (float*)alloc((size_t)EE * 4);
  float* deg   = (float*)alloc((size_t)NN * 4);
  float* x1acc = (float*)alloc((size_t)NN * 128 * 4);
  float* x2acc = (float*)alloc((size_t)NN * 256 * 4);
  float* emacc = (float*)alloc((size_t)NN * 256 * 4);
  float* skip32= (float*)alloc((size_t)NN * 256 * 4);
  float* gate32= (float*)alloc((size_t)NN * 256 * 4);
  float* x2f   = (float*)alloc((size_t)NN * 256 * 4);
  // f16 transposed weights
  h16* tCls1 = (h16*)alloc((size_t)64 * 64 * 2);
  h16* tC11  = (h16*)alloc((size_t)512 * 128 * 2);
  h16* tC12  = (h16*)alloc((size_t)128 * 128 * 2);
  h16* tC13  = (h16*)alloc((size_t)128 * 128 * 2);
  h16* tC21  = (h16*)alloc((size_t)512 * 256 * 2);
  h16* tC22  = (h16*)alloc((size_t)256 * 256 * 2);
  h16* tC23  = (h16*)alloc((size_t)256 * 256 * 2);
  h16* tE1   = (h16*)alloc((size_t)64 * 256 * 2);
  h16* tE2   = (h16*)alloc((size_t)256 * 512 * 2);
  h16* tE3   = (h16*)alloc((size_t)512 * 256 * 2);
  h16* tGate = (h16*)alloc((size_t)256 * 256 * 2);
  h16* tNp1  = (h16*)alloc((size_t)512 * 256 * 2);
  h16* tNp2  = (h16*)alloc((size_t)256 * 128 * 2);
  h16* tSkip = (h16*)alloc((size_t)128 * 256 * 2);

  auto cvt = [&](const float* W, h16* Wt, int K, int N) {
    int n = K * N;
    k_convert_weight_t<<<(n + 255) / 256, 256, 0, stream>>>(W, Wt, K, N);
  };
  cvt(cls1W, tCls1, 64, 64);
  cvt(c11W, tC11, 512, 128); cvt(c12W, tC12, 128, 128); cvt(c13W, tC13, 128, 128);
  cvt(c21W, tC21, 512, 256); cvt(c22W, tC22, 256, 256); cvt(c23W, tC23, 256, 256);
  cvt(enc1W, tE1, 64, 256);  cvt(enc2W, tE2, 256, 512); cvt(enc3W, tE3, 512, 256);
  cvt(gateW, tGate, 256, 256);
  cvt(np1W, tNp1, 512, 256); cvt(np2W, tNp2, 256, 128);
  cvt(skipW, tSkip, 128, 256);

  auto gemm = [&](const h16* A, const h16* Wt, const float* bias,
                  float* Cf, h16* Ch, int M, int N, int K, int act) {
    dim3 g((M + 127) / 128, N / 64);
    k_gemm_wmma<<<g, 256, 0, stream>>>(A, Wt, bias, Cf, Ch, M, N, K, act);
  };

  for (int b = 0; b < BB; ++b) {
    const float* xb = x_in + (size_t)b * NN * 128;
    const float* eab = ea + (size_t)b * EE * 64;
    const int* src = ei + (size_t)b * 2 * EE;
    const int* dst = src + EE;
    float* xfc_b = out + (size_t)b * NN * 512;
    float* p_b = p_base + (size_t)b * NN;

    // zero accumulators
    k_zero<<<(NN + 255) / 256, 256, 0, stream>>>(deg, NN);
    k_zero<<<(NN * 128 + 255) / 256, 256, 0, stream>>>(x1acc, NN * 128);
    k_zero<<<(NN * 256 + 255) / 256, 256, 0, stream>>>(x2acc, NN * 256);
    k_zero<<<(NN * 256 + 255) / 256, 256, 0, stream>>>(emacc, NN * 256);

    // node LN + edge classifier + edge encoder
    k_layernorm_f16<<<(NN + 7) / 8, 256, 0, stream>>>(xb, ln0g, ln0b, xln16, NN, 128);
    k_f32_to_f16<<<(EE * 64 + 255) / 256, 256, 0, stream>>>(eab, ea16, EE * 64);
    gemm(ea16, tCls1, cls1b, nullptr, ecls, EE, 64, 64, 1);                 // relu
    k_rowdot<<<(EE + 255) / 256, 256, 0, stream>>>(ecls, cls2W, cls2b, wcls, EE, 64, 1);
    k_layernorm_f16<<<(EE + 7) / 8, 256, 0, stream>>>(eab, elng, elnb, ealn16, EE, 64);
    gemm(ealn16, tE1, enc1b, nullptr, e1h, EE, 256, 64, 1);                 // relu
    gemm(e1h, tE2, enc2b, nullptr, mbuf, EE, 512, 256, 1);                  // relu
    gemm(mbuf, tE3, enc3b, F1, nullptr, EE, 256, 512, 0);                   // none
    k_mul_w<<<(EE * 256 + 255) / 256, 256, 0, stream>>>(F1, eench, wcls, EE, 256);

    // degree + edge-feature mean accumulation
    k_deg<<<(EE + 255) / 256, 256, 0, stream>>>(dst, deg, EE);
    k_scatter_add<<<(EE * 256 + 255) / 256, 256, 0, stream>>>(F1, dst, emacc, EE, 256);

    // conv1: gather -> 3-layer MLP -> segment mean -> LN -> leaky_relu
    k_gather_concat<<<(EE * 64 + 255) / 256, 256, 0, stream>>>(xln16, 128, eench, 256,
                                                               src, dst, mbuf, EE);
    gemm(mbuf, tC11, c11b, nullptr, c1h, EE, 128, 512, 1);
    gemm(c1h, tC12, c12b, nullptr, c2h, EE, 128, 128, 1);  // c2h holds [E,128] temp
    gemm(c2h, tC13, c13b, F2, nullptr, EE, 128, 128, 0);
    k_scatter_add<<<(EE * 128 + 255) / 256, 256, 0, stream>>>(F2, dst, x1acc, EE, 128);
    k_mean_ln_act<<<(NN + 7) / 8, 256, 0, stream>>>(x1acc, deg, ln1g, ln1b,
                                                    nullptr, 0, x1h, 128, NN, 128, 4);

    // conv2: gather -> 3-layer MLP -> segment mean -> LN -> relu
    k_gather_concat<<<(EE * 64 + 255) / 256, 256, 0, stream>>>(x1h, 128, eench, 256,
                                                               src, dst, mbuf, EE);
    gemm(mbuf, tC21, c21b, nullptr, c2h, EE, 256, 512, 1);
    gemm(c2h, tC22, c22b, nullptr, e1h, EE, 256, 256, 1);
    gemm(e1h, tC23, c23b, F2, nullptr, EE, 256, 256, 0);
    k_scatter_add<<<(EE * 256 + 255) / 256, 256, 0, stream>>>(F2, dst, x2acc, EE, 256);
    k_mean_ln_act<<<(NN + 7) / 8, 256, 0, stream>>>(x2acc, deg, ln2g, ln2b,
                                                    x2f, 256, nullptr, 0, NN, 256, 1);

    // gated skip + output assembly
    gemm(xln16, tSkip, skipb, skip32, skiph, NN, 256, 128, 0);
    gemm(skiph, tGate, gateb, gate32, nullptr, NN, 256, 256, 2);            // sigmoid
    k_xf_blend<<<(NN * 256 + 255) / 256, 256, 0, stream>>>(skip32, gate32, x2f,
                                                           xfc_b, 512, xfh, 512, NN, 256);
    k_mean_scale<<<(NN * 256 + 255) / 256, 256, 0, stream>>>(emacc, deg, xfc_b + 256, 512,
                                                             xfh + 256, 512, NN, 256);

    // node prediction head
    gemm(xfh, tNp1, np1b, nullptr, h1h, NN, 256, 512, 3);                   // elu
    gemm(h1h, tNp2, np2b, nullptr, h2h, NN, 128, 256, 3);                   // elu
    k_rowdot<<<(NN + 255) / 256, 256, 0, stream>>>(h2h, np3W, np3b, p_b, NN, 128, 0);
  }
}
